// WanSelfAttention_80487687127159
// MI455X (gfx1250) — matrix operand order repaired and checked by reading, not compiled, not run on hardware
//
#include <hip/hip_runtime.h>

#define L_TOK 3520
#define DIMSZ 1536
#define NHEADS 12
#define HDIM 128
#define EPSV 1e-6f
#define SCALEV 0.08838834764831845f   // 1/sqrt(128)

typedef __bf16 v16bf __attribute__((ext_vector_type(16)));
typedef __bf16 v8bf  __attribute__((ext_vector_type(8)));
typedef float  v8f   __attribute__((ext_vector_type(8)));

static __device__ __forceinline__ v8f wmma_bf16(v16bf a, v16bf b, v8f c) {
  // D = A(16x32) * B(32x16) + C, fp32 accumulate
  return __builtin_amdgcn_wmma_f32_16x16x32_bf16(false, a, false, b, (short)0, c, false, false);
}

// A-operand per-lane load: two contiguous 8-element (16B) chunks at p and p+16,
// matching the ISA 16-bit A 16x32 layout (lane<16: K{0..7,16..23}; lane>=16: K{8..15,24..31}
// selected by the caller via the +8 element base offset).
static __device__ __forceinline__ v16bf load_afrag(const __bf16* p) {
  v8bf lo = *(const v8bf*)p;
  v8bf hi = *(const v8bf*)(p + 16);
  return __builtin_shufflevector(lo, hi, 0,1,2,3,4,5,6,7,8,9,10,11,12,13,14,15);
}

// ---------------- fp32 -> bf16 convert ----------------
__global__ void cvt_bf16_kernel(const float* __restrict__ in, __bf16* __restrict__ out, int n) {
  int i = blockIdx.x * 256 + threadIdx.x;
  if (i < n) out[i] = (__bf16)in[i];
}

// ---------------- GEMM: C[M,N] = A(bf16 MxK) * B(bf16 NxK)^T + bias ----------------
// One wave computes a 32x64 tile: 2 A-frags x 4 B-frags, 8 WMMAs per K-step of 32.
__global__ __launch_bounds__(32)
void gemm_bf16_kernel(const __bf16* __restrict__ A, const __bf16* __restrict__ B,
                      const float* __restrict__ bias, float* __restrict__ C,
                      int M, int N, int K) {
  const int lane = threadIdx.x;
  const int half = lane >> 4;     // 0: lanes 0-15, 1: lanes 16-31
  const int l16  = lane & 15;
  const int rbase = blockIdx.x * 32;
  const int nbase = blockIdx.y * 64;

  v8f acc[2][4];
  #pragma unroll
  for (int mt = 0; mt < 2; mt++)
    #pragma unroll
    for (int nt = 0; nt < 4; nt++)
      acc[mt][nt] = (v8f){0.f,0.f,0.f,0.f,0.f,0.f,0.f,0.f};

  for (int k = 0; k < K; k += 32) {
    v16bf a0 = load_afrag(A + (size_t)(rbase +      l16) * K + k + half * 8);
    v16bf a1 = load_afrag(A + (size_t)(rbase + 16 + l16) * K + k + half * 8);
    #pragma unroll
    for (int nt = 0; nt < 4; nt++) {
      // B[k][n] = W[n][k]: lane n holds K{0..7,16..23} (lane<16) / K{8..15,24..31} (lane>=16) from row-major W
      v16bf b = load_afrag(B + (size_t)(nbase + nt * 16 + l16) * K + k + half * 8);
      acc[0][nt] = wmma_bf16(a0, b, acc[0][nt]);
      acc[1][nt] = wmma_bf16(a1, b, acc[1][nt]);
    }
  }

  #pragma unroll
  for (int nt = 0; nt < 4; nt++) {
    const int col = nbase + nt * 16 + l16;
    const float bv = bias[col];
    #pragma unroll
    for (int mt = 0; mt < 2; mt++)
      #pragma unroll
      for (int r = 0; r < 8; r++) {
        const int m = rbase + mt * 16 + half * 8 + r;   // C/D layout: row = (lane/16)*8 + vgpr
        C[(size_t)m * N + col] = acc[mt][nt][r] + bv;
      }
  }
}

// ---------------- RMSNorm + RoPE epilogue (q/k), fp32 -> bf16 ----------------
__global__ __launch_bounds__(256)
void norm_rope_kernel(const float* __restrict__ c, const float* __restrict__ g,
                      const int* __restrict__ grid_sizes, const float* __restrict__ freqs,
                      __bf16* __restrict__ out) {
  const int token = blockIdx.x;
  const int tid = threadIdx.x;
  const float* row = c + (size_t)token * DIMSZ;

  float x[6];
  float local = 0.f;
  #pragma unroll
  for (int i = 0; i < 6; i++) { x[i] = row[tid * 6 + i]; local += x[i] * x[i]; }

  #pragma unroll
  for (int off = 16; off > 0; off >>= 1) local += __shfl_xor(local, off, 32);
  __shared__ float ws[8];
  if ((tid & 31) == 0) ws[tid >> 5] = local;
  __syncthreads();
  float tot = 0.f;
  #pragma unroll
  for (int i = 0; i < 8; i++) tot += ws[i];
  const float rinv = rsqrtf(tot * (1.0f / DIMSZ) + EPSV);

  const int h = grid_sizes[1], w = grid_sizes[2];
  const int fi = token / (h * w);
  const int hi = (token / w) % h;
  const int wi = token % w;

  #pragma unroll
  for (int pp = 0; pp < 3; pp++) {
    const int p = tid * 3 + pp;       // global pair index; head = p/64, j = p%64
    const int j = p & 63;
    const int pos = (j < 22) ? fi : (j < 43 ? hi : wi);
    const float ang = freqs[pos * 64 + j];
    const float cs = __cosf(ang), sn = __sinf(ang);
    const float x0 = x[pp * 2]     * rinv * g[tid * 6 + pp * 2];
    const float x1 = x[pp * 2 + 1] * rinv * g[tid * 6 + pp * 2 + 1];
    out[(size_t)token * DIMSZ + 2 * p]     = (__bf16)(x0 * cs - x1 * sn);
    out[(size_t)token * DIMSZ + 2 * p + 1] = (__bf16)(x0 * sn + x1 * cs);
  }
}

// ---------------- V transpose: f32 (L, H, D) -> bf16 per-head (H, D, L) ----------------
__global__ void vtrans_kernel(const float* __restrict__ v, __bf16* __restrict__ vt, int n) {
  int idx = blockIdx.x * 256 + threadIdx.x;
  if (idx >= n) return;
  const int col = idx % DIMSZ;
  const int token = idx / DIMSZ;
  const int head = col >> 7;
  const int d = col & 127;
  vt[((size_t)head * HDIM + d) * L_TOK + token] = (__bf16)v[idx];
}

// ---------------- Flash attention: one wave = (head, 16-row Q tile) ----------------
// Computes S^T = K*Q^T so the D-register layout of S^T matches the A-operand layout
// of P for the P*V WMMA -- no LDS / shuffle transpose needed.
__global__ __launch_bounds__(32)
void attn_kernel(const __bf16* __restrict__ qb, const __bf16* __restrict__ kb,
                 const __bf16* __restrict__ vt, __bf16* __restrict__ ab) {
  const int lane = threadIdx.x;
  const int half = lane >> 4;
  const int l16  = lane & 15;
  const int qbase = blockIdx.x * 16;
  const int head  = blockIdx.y;
  const size_t hoff = (size_t)head * HDIM;

  // Q as B-operand of S^T (B[d][qcol] = q[qcol][d]); held for whole kernel
  v16bf qf[4];
  #pragma unroll
  for (int ds = 0; ds < 4; ds++)
    qf[ds] = load_afrag(qb + (size_t)(qbase + l16) * DIMSZ + hoff + ds * 32 + half * 8);

  v8f o[8];
  #pragma unroll
  for (int t = 0; t < 8; t++) o[t] = (v8f){0.f,0.f,0.f,0.f,0.f,0.f,0.f,0.f};
  float m_run = -3.0e38f;
  float l_run = 0.f;

  for (int kc = 0; kc < L_TOK; kc += 32) {
    v8f s0 = (v8f){0.f,0.f,0.f,0.f,0.f,0.f,0.f,0.f};
    v8f s1 = (v8f){0.f,0.f,0.f,0.f,0.f,0.f,0.f,0.f};
    #pragma unroll
    for (int ds = 0; ds < 4; ds++) {
      const __bf16* kp0 = kb + (size_t)(kc + l16) * DIMSZ + hoff + ds * 32 + half * 8;
      v16bf a0 = load_afrag(kp0);             // keys kc..kc+15 as A rows
      v16bf a1 = load_afrag(kp0 + 16 * DIMSZ); // keys kc+16..kc+31
      s0 = wmma_bf16(a0, qf[ds], s0);
      s1 = wmma_bf16(a1, qf[ds], s1);
    }

    // online softmax over this 32-key chunk (per-lane row = lane&15, dup in both halves)
    float vmax = -3.0e38f;
    #pragma unroll
    for (int c = 0; c < 8; c++) { vmax = fmaxf(vmax, s0[c]); vmax = fmaxf(vmax, s1[c]); }
    vmax = fmaxf(vmax, __shfl_xor(vmax, 16, 32));
    const float m_new = fmaxf(m_run, vmax * SCALEV);
    const float alpha = __expf(m_run - m_new);

    v8f p0, p1;
    float rs = 0.f;
    #pragma unroll
    for (int c = 0; c < 8; c++) {
      p0[c] = __expf(s0[c] * SCALEV - m_new); rs += p0[c];
      p1[c] = __expf(s1[c] * SCALEV - m_new); rs += p1[c];
    }
    rs += __shfl_xor(rs, 16, 32);
    l_run = l_run * alpha + rs;
    m_run = m_new;

    // rescale O rows (row m = (lane/16)*8 + r; alpha for row j lives in lane j)
    const int sb = half * 8;
    #pragma unroll
    for (int r = 0; r < 8; r++) {
      const float ar = __shfl(alpha, sb + r, 32);
      #pragma unroll
      for (int t = 0; t < 8; t++) o[t][r] *= ar;
    }

    // P as A-operand: direct register repack of the two S^T tiles (the key trick)
    v16bf pf;
    #pragma unroll
    for (int c = 0; c < 8; c++) { pf[c] = (__bf16)p0[c]; pf[8 + c] = (__bf16)p1[c]; }

    #pragma unroll
    for (int t = 0; t < 8; t++) {
      // B[k][n] = V[key][d] from pre-transposed V^T: keys {0..7,16..23} (lane<16) / {8..15,24..31} (lane>=16)
      v16bf bf = load_afrag(vt + (hoff + t * 16 + l16) * L_TOK + kc + half * 8);
      o[t] = wmma_bf16(pf, bf, o[t]);
    }
  }

  const int sb = half * 8;
  #pragma unroll
  for (int r = 0; r < 8; r++) {
    const float lr = __shfl(l_run, sb + r, 32);
    const float inv = 1.0f / lr;
    const int token = qbase + sb + r;
    #pragma unroll
    for (int t = 0; t < 8; t++)
      ab[(size_t)token * DIMSZ + hoff + t * 16 + l16] = (__bf16)(o[t][r] * inv);
  }
}

extern "C" void kernel_launch(void* const* d_in, const int* in_sizes, int n_in,
                              void* d_out, int out_size, void* d_ws, size_t ws_size,
                              hipStream_t stream) {
  const float* x          = (const float*)d_in[0];
  const int*   grid_sizes = (const int*)d_in[2];
  const float* freqs      = (const float*)d_in[3];
  const float* Wq = (const float*)d_in[4];  const float* bq = (const float*)d_in[5];
  const float* Wk = (const float*)d_in[6];  const float* bk = (const float*)d_in[7];
  const float* Wv = (const float*)d_in[8];  const float* bv = (const float*)d_in[9];
  const float* Wo = (const float*)d_in[10]; const float* bo = (const float*)d_in[11];
  const float* gq = (const float*)d_in[12]; const float* gk = (const float*)d_in[13];
  float* out = (float*)d_out;

  char* ws = (char*)d_ws;
  size_t off = 0;
  auto alloc = [&](size_t bytes) -> void* {
    void* p = ws + off;
    off += (bytes + 255) & ~(size_t)255;
    return p;
  };
  const size_t NX = (size_t)L_TOK * DIMSZ;          // 5,406,720
  const size_t NW = (size_t)DIMSZ * DIMSZ;          // 2,359,296

  __bf16* xb  = (__bf16*)alloc(NX * 2);
  __bf16* wqb = (__bf16*)alloc(NW * 2);
  __bf16* wkb = (__bf16*)alloc(NW * 2);
  __bf16* wvb = (__bf16*)alloc(NW * 2);
  __bf16* wob = (__bf16*)alloc(NW * 2);
  float*  cq  = (float*)alloc(NX * 4);
  float*  ck  = (float*)alloc(NX * 4);
  float*  cv  = (float*)alloc(NX * 4);
  __bf16* qb  = (__bf16*)alloc(NX * 2);
  __bf16* kb  = (__bf16*)alloc(NX * 2);
  __bf16* vt  = (__bf16*)alloc(NX * 2);
  __bf16* ab  = (__bf16*)alloc(NX * 2);

  // 1) fp32 -> bf16
  cvt_bf16_kernel<<<(int)((NX + 255) / 256), 256, 0, stream>>>(x,  xb,  (int)NX);
  cvt_bf16_kernel<<<(int)((NW + 255) / 256), 256, 0, stream>>>(Wq, wqb, (int)NW);
  cvt_bf16_kernel<<<(int)((NW + 255) / 256), 256, 0, stream>>>(Wk, wkb, (int)NW);
  cvt_bf16_kernel<<<(int)((NW + 255) / 256), 256, 0, stream>>>(Wv, wvb, (int)NW);
  cvt_bf16_kernel<<<(int)((NW + 255) / 256), 256, 0, stream>>>(Wo, wob, (int)NW);

  // 2) QKV projections
  dim3 ggrid(L_TOK / 32, DIMSZ / 64);
  gemm_bf16_kernel<<<ggrid, 32, 0, stream>>>(xb, wqb, bq, cq, L_TOK, DIMSZ, DIMSZ);
  gemm_bf16_kernel<<<ggrid, 32, 0, stream>>>(xb, wkb, bk, ck, L_TOK, DIMSZ, DIMSZ);
  gemm_bf16_kernel<<<ggrid, 32, 0, stream>>>(xb, wvb, bv, cv, L_TOK, DIMSZ, DIMSZ);

  // 3) RMSNorm + RoPE on q/k; V transpose to per-head (D, L)
  norm_rope_kernel<<<L_TOK, 256, 0, stream>>>(cq, gq, grid_sizes, freqs, qb);
  norm_rope_kernel<<<L_TOK, 256, 0, stream>>>(ck, gk, grid_sizes, freqs, kb);
  vtrans_kernel<<<(int)((NX + 255) / 256), 256, 0, stream>>>(cv, vt, (int)NX);

  // 4) Flash attention
  dim3 agrid(L_TOK / 16, NHEADS);
  attn_kernel<<<agrid, 32, 0, stream>>>(qb, kb, vt, ab);

  // 5) Output projection -> d_out (fp32)
  gemm_bf16_kernel<<<ggrid, 32, 0, stream>>>(ab, wob, bo, out, L_TOK, DIMSZ, DIMSZ);
}
